// AttentionBlock_52991306498532
// MI455X (gfx1250) — matrix operand order repaired and checked by reading, not compiled, not run on hardware
//
#include <hip/hip_runtime.h>
#include <hip/hip_bf16.h>

// ---------------------------------------------------------------------------
// AttentionBlock for MI455X (gfx1250): bf16 WMMA + double-buffered async
// global->LDS staging (GLOBAL_LOAD_ASYNC_TO_LDS_B128 / ASYNCcnt pipelining).
//   B=4, S=2048, D=1024, H=16, DH=64, T=B*S=8192
// ---------------------------------------------------------------------------

typedef __attribute__((ext_vector_type(16))) __bf16        v16bf;
typedef __attribute__((ext_vector_type(8)))  float         v8f;
typedef __attribute__((ext_vector_type(4)))  unsigned int  v4u;

// vector-of-4-int pointer types for the async global->LDS builtin
typedef int v4i_ __attribute__((vector_size(16)));
typedef __attribute__((address_space(1))) v4i_ g_v4i;   // global
typedef __attribute__((address_space(3))) v4i_ l_v4i;   // LDS

union Frag16 { v16bf bf; v4u q[2]; };

#define B_  4
#define S_  2048
#define D_  1024
#define H_  16
#define DH_ 64
#define T_  (B_ * S_)      // 8192
#define N3_ (3 * H_ * DH_) // 3072

#if defined(__has_builtin)
#if __has_builtin(__builtin_amdgcn_global_load_async_to_lds_b128)
#define HAVE_ASYNC_LDS 1
#endif
#endif

__device__ __forceinline__ void cp16(unsigned short* dst_lds,
                                     const unsigned short* src_glb) {
#if defined(HAVE_ASYNC_LDS)
  __builtin_amdgcn_global_load_async_to_lds_b128(
      (g_v4i*)src_glb, (l_v4i*)dst_lds, 0, 0);
#else
  *(v4u*)dst_lds = *(const v4u*)src_glb;
#endif
}

__device__ __forceinline__ void cp_wait0() {
#if defined(HAVE_ASYNC_LDS)
#if __has_builtin(__builtin_amdgcn_s_wait_asynccnt)
  __builtin_amdgcn_s_wait_asynccnt(0);
#else
  asm volatile("s_wait_asynccnt 0" ::: "memory");
#endif
#endif
}
__device__ __forceinline__ void cp_wait4() {
#if defined(HAVE_ASYNC_LDS)
#if __has_builtin(__builtin_amdgcn_s_wait_asynccnt)
  __builtin_amdgcn_s_wait_asynccnt(4);
#else
  asm volatile("s_wait_asynccnt 4" ::: "memory");
#endif
#endif
}
__device__ __forceinline__ void cp_wait6() {
#if defined(HAVE_ASYNC_LDS)
#if __has_builtin(__builtin_amdgcn_s_wait_asynccnt)
  __builtin_amdgcn_s_wait_asynccnt(6);
#else
  asm volatile("s_wait_asynccnt 6" ::: "memory");
#endif
#endif
}

__device__ __forceinline__ unsigned short f2bf(float f) {
  unsigned int u = __float_as_uint(f);
  unsigned int r = u + 0x7FFFu + ((u >> 16) & 1u); // round-to-nearest-even
  return (unsigned short)(r >> 16);
}

// A-fragment (16x32 bf16), row-major source with row stride ld
__device__ __forceinline__ v16bf load_a16(const unsigned short* A, int ld, int k0) {
  int lane = threadIdx.x & 31;
  int row  = lane & 15;
  int kb   = (lane >> 4) * 8;
  const unsigned short* p = A + row * ld + k0 + kb;
  Frag16 f;
  f.q[0] = *(const v4u*)(p);
  f.q[1] = *(const v4u*)(p + 16);
  return f.bf;
}

// B-fragment (32x16 bf16) from column-major storage Bt[N][K], row stride ld
__device__ __forceinline__ v16bf load_bt16(const unsigned short* Bt, int ld, int k0) {
  int lane = threadIdx.x & 31;
  int col  = lane & 15;
  int kb   = (lane >> 4) * 8;
  const unsigned short* p = Bt + col * ld + k0 + kb;
  Frag16 f;
  f.q[0] = *(const v4u*)(p);
  f.q[1] = *(const v4u*)(p + 16);
  return f.bf;
}

__device__ __forceinline__ v8f wmma_bf16(v16bf a, v16bf b, v8f c) {
  return __builtin_amdgcn_wmma_f32_16x16x32_bf16(
      false, a, false, b, (short)0, c, false, false);
}

// ---------------------------------------------------------------------------
// Kernel 1: convert / transpose weights+activations to bf16
// ---------------------------------------------------------------------------
__global__ void convert_k(const float* __restrict__ x,
                          const float* __restrict__ wqkv,
                          const float* __restrict__ wout,
                          unsigned short* __restrict__ Xb,
                          unsigned short* __restrict__ Wqkvt,
                          unsigned short* __restrict__ Woutt) {
  const int N1 = T_ * D_;
  const int N2 = D_ * N3_;
  const int N3 = D_ * D_;
  const int total = N1 + N2 + N3;
  for (int i = blockIdx.x * blockDim.x + threadIdx.x; i < total;
       i += gridDim.x * blockDim.x) {
    if (i < N1) {
      Xb[i] = f2bf(x[i]);
    } else if (i < N1 + N2) {
      int j = i - N1;
      int k = j / N3_, n = j % N3_;
      Wqkvt[n * D_ + k] = f2bf(wqkv[j]);
    } else {
      int j = i - N1 - N2;
      int k = j >> 10, n = j & 1023;
      Woutt[n * D_ + k] = f2bf(wout[j]);
    }
  }
}

// ---------------------------------------------------------------------------
// Kernel 2: QKV GEMM (M=8192, N=3072, K=1024)
// 256 thr = 8 waves (2x4), block tile 64x128, wave tile 32x32,
// k-stage = 64 deep, double-buffered async LDS staging (6 ops/thread/stage).
// ---------------------------------------------------------------------------
__global__ __launch_bounds__(256) void qkv_gemm(
    const unsigned short* __restrict__ Xb,
    const unsigned short* __restrict__ Wt,
    const float* __restrict__ bias,
    unsigned short* __restrict__ Qb,
    unsigned short* __restrict__ Kb,
    unsigned short* __restrict__ Vt) {
  __shared__ unsigned short At[2][64 * 64];    // 2 x  8 KiB
  __shared__ unsigned short Bt[2][128 * 64];   // 2 x 16 KiB
  int tid = threadIdx.x;
  int w = tid >> 5, lane = tid & 31;
  int wm = w >> 2, wn = w & 3;
  const int NB = N3_ / 128;                    // 24
  int bm = blockIdx.x / NB;
  int bn = blockIdx.x % NB;
  const unsigned short* Ab = Xb + bm * 64 * D_;
  const unsigned short* Bb = Wt + bn * 128 * D_;

  auto stage = [&](int buf, int k0) {
#pragma unroll
    for (int i = 0; i < 2; ++i) {              // A: 512 chunks
      int cid = tid + i * 256;
      int row = cid >> 3, ch = cid & 7;
      cp16(&At[buf][row * 64 + ch * 8], Ab + row * D_ + k0 + ch * 8);
    }
#pragma unroll
    for (int i = 0; i < 4; ++i) {              // B: 1024 chunks
      int cid = tid + i * 256;
      int row = cid >> 3, ch = cid & 7;
      cp16(&Bt[buf][row * 64 + ch * 8], Bb + row * D_ + k0 + ch * 8);
    }
  };

  v8f c00 = {}, c01 = {}, c10 = {}, c11 = {};
  const int NK = D_ / 64;                      // 16 stages
  stage(0, 0);
  for (int s = 0; s < NK; ++s) {
    int cb = s & 1;
    __syncthreads();                           // old contents of buf^1 free
    if (s + 1 < NK) { stage(cb ^ 1, (s + 1) * 64); cp_wait6(); }
    else            { cp_wait0(); }
    __syncthreads();                           // buf[cb] ready for all waves
    const unsigned short* Acb = At[cb];
    const unsigned short* Bcb = Bt[cb];
#pragma unroll
    for (int kk = 0; kk < 64; kk += 32) {
      v16bf a0 = load_a16(&Acb[(wm * 32) * 64], 64, kk);
      v16bf a1 = load_a16(&Acb[(wm * 32 + 16) * 64], 64, kk);
      v16bf b0 = load_bt16(&Bcb[(wn * 32) * 64], 64, kk);
      v16bf b1 = load_bt16(&Bcb[(wn * 32 + 16) * 64], 64, kk);
      c00 = wmma_bf16(a0, b0, c00);
      c01 = wmma_bf16(a0, b1, c01);
      c10 = wmma_bf16(a1, b0, c10);
      c11 = wmma_bf16(a1, b1, c11);
    }
  }

  int hl = lane >> 4, nl = lane & 15;
#pragma unroll
  for (int mi = 0; mi < 2; ++mi) {
#pragma unroll
    for (int ni = 0; ni < 2; ++ni) {
      v8f c = (mi == 0) ? (ni == 0 ? c00 : c01) : (ni == 0 ? c10 : c11);
      int m0 = bm * 64 + wm * 32 + mi * 16;
      int n0 = bn * 128 + wn * 32 + ni * 16;
      int t3  = n0 >> 10;
      int rem = n0 & 1023;
      int h   = rem >> 6;
      int dh  = (rem & 63) + nl;
      float bv = bias[t3 * (H_ * DH_) + h * DH_ + dh];
#pragma unroll
      for (int r = 0; r < 8; ++r) {
        int token = m0 + r + hl * 8;
        int bb = token >> 11;
        int ss = token & (S_ - 1);
        float v = c[r] + bv;
        if (t3 == 0) {
          Qb[((bb * H_ + h) * S_ + ss) * DH_ + dh] = f2bf(v * 0.125f);
        } else if (t3 == 1) {
          Kb[((bb * H_ + h) * S_ + ss) * DH_ + dh] = f2bf(v);
        } else {
          Vt[((bb * H_ + h) * DH_ + dh) * S_ + ss] = f2bf(v);
        }
      }
    }
  }
}

// ---------------------------------------------------------------------------
// Kernel 3: causal flash attention, 64 q-rows/block (16 per wave),
// double-buffered K/V tiles (32 keys/stage) shared by 4 waves.
// ---------------------------------------------------------------------------
__global__ __launch_bounds__(128) void flash_attn(
    const unsigned short* __restrict__ Qb,
    const unsigned short* __restrict__ Kb,
    const unsigned short* __restrict__ Vt,
    unsigned short* __restrict__ Ob) {
  __shared__ unsigned short Kt[2][32 * 64];     // 2 x 4 KiB  [key][dh]
  __shared__ unsigned short Vtt[2][64 * 32];    // 2 x 4 KiB  [dh][key]
  __shared__ unsigned short pbuf[4][16 * 32];   // 4 KiB      P tiles
  int tid = threadIdx.x;
  int w = tid >> 5, lane = tid & 31;
  int hl = lane >> 4, nl = lane & 15;

  int qt64 = blockIdx.x & (S_ / 64 - 1);
  int h    = (blockIdx.x >> 5) & (H_ - 1);
  int b    = blockIdx.x >> 9;
  int qblk = qt64 * 64;
  int qw   = qblk + w * 16;

  const unsigned short* Qh = Qb + (b * H_ + h) * S_ * DH_;
  const unsigned short* Kh = Kb + (b * H_ + h) * S_ * DH_;
  const unsigned short* Vh = Vt + (b * H_ + h) * DH_ * S_;

  v16bf aq0 = load_a16(Qh + qw * DH_, DH_, 0);
  v16bf aq1 = load_a16(Qh + qw * DH_, DH_, 32);

  v8f acc0 = {}, acc1 = {}, acc2 = {}, acc3 = {};
  float mrun[8], lrun[8];
#pragma unroll
  for (int r = 0; r < 8; ++r) { mrun[r] = -1e30f; lrun[r] = 0.0f; }

  auto stage = [&](int buf, int k0) {
#pragma unroll
    for (int i = 0; i < 2; ++i) {               // K: 32x64 = 256 chunks
      int cid = tid + i * 128;
      int row = cid >> 3, ch = cid & 7;
      cp16(&Kt[buf][row * 64 + ch * 8], Kh + (k0 + row) * DH_ + ch * 8);
    }
#pragma unroll
    for (int i = 0; i < 2; ++i) {               // V^T: 64x32 = 256 chunks
      int cid = tid + i * 128;
      int row = cid >> 2, ch = cid & 3;
      cp16(&Vtt[buf][row * 32 + ch * 8], Vh + row * S_ + k0 + ch * 8);
    }
  };

  unsigned short* pb = pbuf[w];
  int q_hi_w = qw + 15;
  int nstage = qblk / 32 + 2;                   // stages of 32 keys
  stage(0, 0);
  for (int s = 0; s < nstage; ++s) {
    int k0 = s * 32;
    int cb = s & 1;
    __syncthreads();
    if (s + 1 < nstage) { stage(cb ^ 1, k0 + 32); cp_wait4(); }
    else                { cp_wait0(); }
    __syncthreads();

    if (k0 <= q_hi_w) {                         // wave not fully masked
      const unsigned short* Kcb = Kt[cb];
      const unsigned short* Vcb = Vtt[cb];
      v8f s0 = {}, s1 = {};
      s0 = wmma_bf16(aq0, load_bt16(Kcb, 64, 0), s0);
      s0 = wmma_bf16(aq1, load_bt16(Kcb, 64, 32), s0);
      s1 = wmma_bf16(aq0, load_bt16(&Kcb[16 * 64], 64, 0), s1);
      s1 = wmma_bf16(aq1, load_bt16(&Kcb[16 * 64], 64, 32), s1);

#pragma unroll
      for (int r = 0; r < 8; ++r) {
        int m = r + hl * 8;
        int qrow = qw + m;
        float v0 = ((k0 + nl) > qrow) ? -1e30f : s0[r];
        float v1 = ((k0 + 16 + nl) > qrow) ? -1e30f : s1[r];
        float t = fmaxf(v0, v1);
        t = fmaxf(t, __shfl_xor(t, 1));
        t = fmaxf(t, __shfl_xor(t, 2));
        t = fmaxf(t, __shfl_xor(t, 4));
        t = fmaxf(t, __shfl_xor(t, 8));
        float nm = fmaxf(mrun[r], t);
        float alpha = __expf(mrun[r] - nm);
        float p0 = __expf(v0 - nm);
        float p1 = __expf(v1 - nm);
        float rs = p0 + p1;
        rs += __shfl_xor(rs, 1);
        rs += __shfl_xor(rs, 2);
        rs += __shfl_xor(rs, 4);
        rs += __shfl_xor(rs, 8);
        lrun[r] = lrun[r] * alpha + rs;
        mrun[r] = nm;
        acc0[r] *= alpha; acc1[r] *= alpha; acc2[r] *= alpha; acc3[r] *= alpha;
        pb[m * 32 + nl]      = f2bf(p0);
        pb[m * 32 + 16 + nl] = f2bf(p1);
      }
      v16bf ap = load_a16(pb, 32, 0);           // P tile as A fragment

      acc0 = wmma_bf16(ap, load_bt16(&Vcb[0 * 16 * 32], 32, 0), acc0);
      acc1 = wmma_bf16(ap, load_bt16(&Vcb[1 * 16 * 32], 32, 0), acc1);
      acc2 = wmma_bf16(ap, load_bt16(&Vcb[2 * 16 * 32], 32, 0), acc2);
      acc3 = wmma_bf16(ap, load_bt16(&Vcb[3 * 16 * 32], 32, 0), acc3);
    }
  }

#pragma unroll
  for (int r = 0; r < 8; ++r) {
    int m = r + hl * 8;
    int token = b * S_ + qw + m;
    float inv = 1.0f / lrun[r];
    unsigned short* o = Ob + token * (H_ * DH_) + h * DH_ + nl;
    o[0]  = f2bf(acc0[r] * inv);
    o[16] = f2bf(acc1[r] * inv);
    o[32] = f2bf(acc2[r] * inv);
    o[48] = f2bf(acc3[r] * inv);
  }
}

// ---------------------------------------------------------------------------
// Kernel 4: output projection GEMM (M=8192, N=1024, K=1024) -> f32 + bias
// ---------------------------------------------------------------------------
__global__ __launch_bounds__(256) void out_gemm(
    const unsigned short* __restrict__ Ob,
    const unsigned short* __restrict__ Wt,
    const float* __restrict__ bias,
    float* __restrict__ Out) {
  __shared__ unsigned short At[2][64 * 64];
  __shared__ unsigned short Bt[2][128 * 64];
  int tid = threadIdx.x;
  int w = tid >> 5, lane = tid & 31;
  int wm = w >> 2, wn = w & 3;
  const int NB = D_ / 128;                     // 8
  int bm = blockIdx.x / NB;
  int bn = blockIdx.x % NB;
  const unsigned short* Ab = Ob + bm * 64 * D_;
  const unsigned short* Bb = Wt + bn * 128 * D_;

  auto stage = [&](int buf, int k0) {
#pragma unroll
    for (int i = 0; i < 2; ++i) {
      int cid = tid + i * 256;
      int row = cid >> 3, ch = cid & 7;
      cp16(&At[buf][row * 64 + ch * 8], Ab + row * D_ + k0 + ch * 8);
    }
#pragma unroll
    for (int i = 0; i < 4; ++i) {
      int cid = tid + i * 256;
      int row = cid >> 3, ch = cid & 7;
      cp16(&Bt[buf][row * 64 + ch * 8], Bb + row * D_ + k0 + ch * 8);
    }
  };

  v8f c00 = {}, c01 = {}, c10 = {}, c11 = {};
  const int NK = D_ / 64;
  stage(0, 0);
  for (int s = 0; s < NK; ++s) {
    int cb = s & 1;
    __syncthreads();
    if (s + 1 < NK) { stage(cb ^ 1, (s + 1) * 64); cp_wait6(); }
    else            { cp_wait0(); }
    __syncthreads();
    const unsigned short* Acb = At[cb];
    const unsigned short* Bcb = Bt[cb];
#pragma unroll
    for (int kk = 0; kk < 64; kk += 32) {
      v16bf a0 = load_a16(&Acb[(wm * 32) * 64], 64, kk);
      v16bf a1 = load_a16(&Acb[(wm * 32 + 16) * 64], 64, kk);
      v16bf b0 = load_bt16(&Bcb[(wn * 32) * 64], 64, kk);
      v16bf b1 = load_bt16(&Bcb[(wn * 32 + 16) * 64], 64, kk);
      c00 = wmma_bf16(a0, b0, c00);
      c01 = wmma_bf16(a0, b1, c01);
      c10 = wmma_bf16(a1, b0, c10);
      c11 = wmma_bf16(a1, b1, c11);
    }
  }

  int hl = lane >> 4, nl = lane & 15;
#pragma unroll
  for (int mi = 0; mi < 2; ++mi) {
#pragma unroll
    for (int ni = 0; ni < 2; ++ni) {
      v8f c = (mi == 0) ? (ni == 0 ? c00 : c01) : (ni == 0 ? c10 : c11);
      int m0 = bm * 64 + wm * 32 + mi * 16;
      int n  = bn * 128 + wn * 32 + ni * 16 + nl;
      float bv = bias[n];
#pragma unroll
      for (int r = 0; r < 8; ++r) {
        int token = m0 + r + hl * 8;
        Out[token * D_ + n] = c[r] + bv;
      }
    }
  }
}

// ---------------------------------------------------------------------------
extern "C" void kernel_launch(void* const* d_in, const int* in_sizes, int n_in,
                              void* d_out, int out_size, void* d_ws, size_t ws_size,
                              hipStream_t stream) {
  const float* x     = (const float*)d_in[0];
  const float* w_qkv = (const float*)d_in[1];
  const float* b_qkv = (const float*)d_in[2];
  const float* w_out = (const float*)d_in[3];
  const float* b_out = (const float*)d_in[4];
  float* out = (float*)d_out;

  char* ws = (char*)d_ws;
  unsigned short* Xb    = (unsigned short*)(ws);                      // 16 MiB
  unsigned short* Wqkvt = (unsigned short*)(ws + (size_t)16777216);   //  6 MiB
  unsigned short* Woutt = (unsigned short*)(ws + (size_t)23068672);   //  2 MiB
  unsigned short* Qb    = (unsigned short*)(ws + (size_t)25165824);   // 16 MiB
  unsigned short* Kb    = (unsigned short*)(ws + (size_t)41943040);   // 16 MiB
  unsigned short* Vt    = (unsigned short*)(ws + (size_t)58720256);   // 16 MiB
  unsigned short* Ob    = (unsigned short*)(ws + (size_t)75497472);   // 16 MiB

  convert_k<<<2048, 256, 0, stream>>>(x, w_qkv, w_out, Xb, Wqkvt, Woutt);

  // (8192/64) * (3072/128) = 3072 blocks
  qkv_gemm<<<3072, 256, 0, stream>>>(Xb, Wqkvt, b_qkv, Qb, Kb, Vt);

  // B*H*(S/64) = 2048 blocks
  flash_attn<<<2048, 128, 0, stream>>>(Qb, Kb, Vt, Ob);

  // (8192/64) * (1024/128) = 1024 blocks
  out_gemm<<<1024, 256, 0, stream>>>(Ob, Woutt, b_out, out);
}
